// ContrastiveProtosTeacherLoss_72189810311259
// MI455X (gfx1250) — matrix operand order repaired and checked by reading, not compiled, not run on hardware
//
#include <hip/hip_runtime.h>
#include <hip/hip_bf16.h>

typedef __attribute__((ext_vector_type(16))) _Float16 v16h;
typedef __attribute__((ext_vector_type(8)))  float    v8f;
typedef __attribute__((ext_vector_type(4)))  int      v4i;

#if defined(__HIP_DEVICE_COMPILE__) && __has_builtin(__builtin_amdgcn_global_load_async_to_lds_b128)
#define CPTL_HAVE_ASYNC_LDS 1
#endif

namespace cptl {
constexpr int B  = 2, C = 64, H = 64, W = 64, D = 32;
constexpr int HR = 16, WR = 16, DR = 8;
constexpr int NR = HR * WR * DR;           // 2048 reduced voxels
constexpr int HP = 4, WP = 4, DP = 2;
constexpr int P  = HP * WP * DP;           // 32 protos
constexpr int MT  = NR / 16;               // 128 m-tiles
constexpr int KST = C / 32;                // 2 k-steps of 32
constexpr int NT2 = P / 16;                // 2 n-tiles
constexpr float TEMP_T = 0.033f;
constexpr float TEMP_S = 0.066f;
constexpr float SIG2   = (128.0f / 2.355f) * (128.0f / 2.355f);
constexpr float MAX_D2 = 16.0f;            // (4 mm)^2
} // namespace cptl

using namespace cptl;

__device__ inline void wait_async_zero() {
#if defined(__HIP_DEVICE_COMPILE__)
#if __has_builtin(__builtin_amdgcn_s_wait_asynccnt)
    __builtin_amdgcn_s_wait_asynccnt(0);
#else
    asm volatile("s_wait_asynccnt 0x0" ::: "memory");
#endif
#endif
}

// ---------------- trilinear resize (align_corners=false, single reflection) --
__device__ inline void axis_lin(int i, int n, int out_n, int& i0, int& i1, float& w) {
    float scale = (float)n / (float)out_n;
    float c = ((float)i + 0.5f) * scale - 0.5f;
    if (c < -0.5f) c = -1.0f - c;
    float nf = (float)n;
    if (c > nf - 0.5f) c = 2.0f * nf - 1.0f - c;
    float f = floorf(c);
    i0 = (int)f;
    if (i0 < 0) i0 = 0;
    if (i0 > n - 1) i0 = n - 1;
    i1 = i0 + 1;
    if (i1 > n - 1) i1 = n - 1;
    w = c - f;
    if (w < 0.0f) w = 0.0f;
    if (w > 1.0f) w = 1.0f;
}

__global__ void k_resize(const float* __restrict__ in, float* __restrict__ out,
                         int BC, int h2, int w2, int d2) {
    int idx = blockIdx.x * blockDim.x + threadIdx.x;
    int total = BC * h2 * w2 * d2;
    if (idx >= total) return;
    int dd = idx % d2; int t = idx / d2;
    int ww = t % w2;   t /= w2;
    int hh = t % h2;   int bc = t / h2;

    int h0, h1, w0, w1, d0, d1;
    float fh, fw, fd;
    axis_lin(hh, H, h2, h0, h1, fh);
    axis_lin(ww, W, w2, w0, w1, fw);
    axis_lin(dd, D, d2, d0, d1, fd);

    const float* base = in + (size_t)bc * H * W * D;
    float wh[2] = {1.0f - fh, fh};
    float wwt[2] = {1.0f - fw, fw};
    float wd[2] = {1.0f - fd, fd};
    int ih[2] = {h0, h1}, iw[2] = {w0, w1}, id[2] = {d0, d1};
    float acc = 0.0f;
#pragma unroll
    for (int a = 0; a < 2; a++)
#pragma unroll
        for (int b2 = 0; b2 < 2; b2++)
#pragma unroll
            for (int c2 = 0; c2 < 2; c2++)
                acc += wh[a] * wwt[b2] * wd[c2] *
                       base[((size_t)ih[a] * W + iw[b2]) * D + id[c2]];
    out[idx] = acc;
}

// --------- [B,C,N] channel-major -> [B,N,C] row-major (+L2-normalized copy) --
__global__ void k_transpose_norm(const float* __restrict__ in, // [B,C,N]
                                 float* __restrict__ raw,      // [B,N,C] or null
                                 float* __restrict__ nrm,      // [B,N,C]
                                 int n_total) {
    int idx = blockIdx.x * blockDim.x + threadIdx.x; // b*n_total + n
    if (idx >= B * n_total) return;
    int b = idx / n_total;
    int n = idx - b * n_total;
    const float* src = in + (size_t)b * C * n_total + n;
    float ss = 0.0f;
    for (int c = 0; c < C; c++) {
        float v = src[(size_t)c * n_total];
        ss += v * v;
    }
    float rn = 1.0f / fmaxf(sqrtf(ss), 1e-12f);
    float* rdst = raw ? raw + (size_t)idx * C : nullptr;
    float* ndst = nrm + (size_t)idx * C;
    for (int c = 0; c < C; c++) {
        float v = src[(size_t)c * n_total];
        if (rdst) rdst[c] = v;
        ndst[c] = v * rn;
    }
}

// --------------------- coord transpose [B,3,N] -> [B,N,3] -------------------
__global__ void k_transpose_coords(const float* __restrict__ in, float* __restrict__ out,
                                   int n_total) {
    int idx = blockIdx.x * blockDim.x + threadIdx.x;
    if (idx >= B * n_total) return;
    int b = idx / n_total;
    int n = idx - b * n_total;
    for (int k = 0; k < 3; k++)
        out[(size_t)idx * 3 + k] = in[((size_t)b * 3 + k) * n_total + n];
}

// -------- nearest teacher voxel per student voxel (squared-dist argmin) -----
// Teacher coords (24 KB) staged into LDS via GLOBAL_LOAD_ASYNC_TO_LDS_B128
// (ASYNCcnt path) when the toolchain exposes the gfx1250 builtin.
__global__ void k_nearest(const float* __restrict__ cs_t, const float* __restrict__ ct_t,
                          int* __restrict__ idxc, float* __restrict__ mask) {
    __shared__ float sct[NR * 3]; // 24 KB of 320 KB WGP LDS
    int b = blockIdx.y;
#ifdef CPTL_HAVE_ASYNC_LDS
    {
        constexpr int CHUNKS = (NR * 3 * (int)sizeof(float)) / 16; // 1536 x 16B
        char* gbase = (char*)(ct_t + (size_t)b * NR * 3);
        char* lbase = (char*)sct;
        for (int i = threadIdx.x; i < CHUNKS; i += blockDim.x) {
            __builtin_amdgcn_global_load_async_to_lds_b128(
                (__attribute__((address_space(1))) v4i*)(gbase + (size_t)i * 16),
                (__attribute__((address_space(3))) v4i*)(lbase + (size_t)i * 16),
                0, 0);
        }
        wait_async_zero(); // each wave drains its own ASYNCcnt before barrier
    }
#else
    for (int i = threadIdx.x; i < NR * 3; i += blockDim.x)
        sct[i] = ct_t[(size_t)b * NR * 3 + i];
#endif
    __syncthreads();
    int n = blockIdx.x * blockDim.x + threadIdx.x;
    if (n >= NR) return;
    const float* p = cs_t + ((size_t)b * NR + n) * 3;
    float x = p[0], y = p[1], z = p[2];
    float best = 3.4e38f;
    int bi = 0;
    for (int t = 0; t < NR; t++) {
        float dx = x - sct[t * 3 + 0];
        float dy = y - sct[t * 3 + 1];
        float dz = (z - sct[t * 3 + 2]) * 2.0f; // SCALE_Z
        float d2 = dx * dx + dy * dy + dz * dz;
        if (d2 < best) { best = d2; bi = t; }
    }
    idxc[(size_t)b * NR + n] = bi;
    mask[(size_t)b * NR + n] = (best <= MAX_D2) ? 1.0f : 0.0f;
}

// ---------------------- proto init (normalize emb_p rows) -------------------
__global__ void k_protos_init(const float* __restrict__ emb_p,   // [B,C,P]
                              const float* __restrict__ coord_p, // [B,3,P]
                              float* __restrict__ protos,        // [B,P,C]
                              float* __restrict__ cp) {          // [B,P,3]
    int idx = blockIdx.x * blockDim.x + threadIdx.x;
    if (idx >= B * P) return;
    int b = idx / P;
    int p = idx - b * P;
    float ss = 0.0f;
    for (int c = 0; c < C; c++) {
        float v = emb_p[((size_t)b * C + c) * P + p];
        ss += v * v;
    }
    float rn = 1.0f / fmaxf(sqrtf(ss), 1e-12f);
    for (int c = 0; c < C; c++)
        protos[(size_t)idx * C + c] = emb_p[((size_t)b * C + c) * P + p] * rn;
    for (int k = 0; k < 3; k++)
        cp[(size_t)idx * 3 + k] = coord_p[((size_t)b * 3 + k) * P + p];
}

// ---------------- WMMA operand packing (ISA 7.12.2 layouts) -----------------
// A 16x32 f16: lane<16 holds row M=lane, K(h) = (h&7) + ((h>>3)<<4);
//              lane>=16 holds row M=lane-16, K shifted by +8.
__device__ inline int kmapA(int lane, int h) {
    return (h & 7) + ((h >> 3) << 4) + ((lane >> 4) << 3);
}
// B 32x16 f16: lane<16 holds column N=lane, K(h) = (h&7) + ((h>>3)<<4);
//              lane>=16 holds column N=lane-16, K shifted by +8.
__device__ inline int kmapB(int lane, int h) {
    return (h & 7) + ((h >> 3) << 4) + ((lane >> 4) << 3);
}

__global__ void k_packA(const float* __restrict__ en, // [B,NR,C] normalized
                        _Float16* __restrict__ ap) {  // [B,MT,KST,32,16]
    int idx = blockIdx.x * blockDim.x + threadIdx.x;
    if (idx >= B * MT * KST * 32 * 16) return;
    int h = idx & 15;
    int lane = (idx >> 4) & 31;
    int ks = (idx >> 9) & (KST - 1);
    int mt = (idx >> 10) & (MT - 1);
    int b = idx >> 17;
    int n = mt * 16 + (lane & 15);
    int c = ks * 32 + kmapA(lane, h);
    ap[idx] = (_Float16)en[((size_t)b * NR + n) * C + c];
}

__global__ void k_packB(const float* __restrict__ pr, // [B,P,C] protos
                        _Float16* __restrict__ bp) {  // [B,NT2,KST,32,16]
    int idx = blockIdx.x * blockDim.x + threadIdx.x;
    if (idx >= B * NT2 * KST * 32 * 16) return;
    int h = idx & 15;
    int lane = (idx >> 4) & 31;
    int ks = (idx >> 9) & (KST - 1);
    int nt = (idx >> 10) & (NT2 - 1);
    int b = idx >> 11;
    int p = nt * 16 + (lane & 15);
    int c = ks * 32 + kmapB(lane, h);
    bp[idx] = (_Float16)pr[((size_t)b * P + p) * C + c];
}

// --------- similarity GEMM: logits[b,n,p] = etn[n,:] . protos[p,:] ----------
// one wave per (m-tile, batch); 4 x v_wmma_f32_16x16x32_f16 per wave
__global__ void __launch_bounds__(32)
k_gemm_wmma(const _Float16* __restrict__ ap, const _Float16* __restrict__ bp,
            float* __restrict__ out) { // [B,NR,P]
    int mt = blockIdx.x;
    int b = blockIdx.y;
    int lane = threadIdx.x;
    v8f acc0 = {0.f, 0.f, 0.f, 0.f, 0.f, 0.f, 0.f, 0.f};
    v8f acc1 = acc0;
#pragma unroll
    for (int ks = 0; ks < KST; ks++) {
        v16h a  = *(const v16h*)(ap + ((((size_t)(b * MT + mt) * KST + ks) * 32 + lane) * 16));
        v16h b0 = *(const v16h*)(bp + ((((size_t)(b * NT2 + 0) * KST + ks) * 32 + lane) * 16));
        v16h b1 = *(const v16h*)(bp + ((((size_t)(b * NT2 + 1) * KST + ks) * 32 + lane) * 16));
        acc0 = __builtin_amdgcn_wmma_f32_16x16x32_f16(false, a, false, b0,
                                                      (short)0, acc0, false, false);
        acc1 = __builtin_amdgcn_wmma_f32_16x16x32_f16(false, a, false, b1,
                                                      (short)0, acc1, false, false);
    }
    // C/D layout: VGPR r -> M = r + (lane>=16 ? 8 : 0); N = lane & 15
    int ncol = lane & 15;
    int mbase = (lane >> 4) << 3;
#pragma unroll
    for (int r = 0; r < 8; r++) {
        int m = mt * 16 + r + mbase;
        out[((size_t)b * NR + m) * P + ncol]      = acc0[r];
        out[((size_t)b * NR + m) * P + 16 + ncol] = acc1[r];
    }
}

// --------- a[b,n,p] = softmax_p(logit/T_T) * exp(-dd^2 / (2 sig^2)) ---------
__global__ void k_soft_weight(const float* __restrict__ logits, // [B,NR,P]
                              const float* __restrict__ ct_t,   // [B,NR,3]
                              const float* __restrict__ cp,     // [B,P,3]
                              float* __restrict__ Amat) {       // [B,NR,P]
    int idx = blockIdx.x * blockDim.x + threadIdx.x;
    if (idx >= B * NR) return;
    int b = idx / NR;
    const float* lrow = logits + (size_t)idx * P;
    float l[P];
    float mx = -3.4e38f;
    for (int p = 0; p < P; p++) { l[p] = lrow[p]; mx = fmaxf(mx, l[p]); }
    float sum = 0.0f;
    for (int p = 0; p < P; p++) { l[p] = __expf((l[p] - mx) / TEMP_T); sum += l[p]; }
    float inv = 1.0f / sum;
    float x = ct_t[(size_t)idx * 3 + 0];
    float y = ct_t[(size_t)idx * 3 + 1];
    float z = ct_t[(size_t)idx * 3 + 2];
    float* arow = Amat + (size_t)idx * P;
    for (int p = 0; p < P; p++) {
        const float* cpp = cp + ((size_t)b * P + p) * 3;
        float dx = x - cpp[0];
        float dy = y - cpp[1];
        float dz = (z - cpp[2]) * 2.0f;
        float dd2 = dx * dx + dy * dy + dz * dz;
        float w = __expf(-dd2 / (2.0f * SIG2));
        arow[p] = l[p] * inv * w;
    }
}

// --- per-proto accumulation: pacc[p,c] = sum_n a*etk ; cpacc ; den (no atomics)
__global__ void __launch_bounds__(64)
k_accum(const float* __restrict__ Amat, const float* __restrict__ etk_t,
        const float* __restrict__ ct_t, float* __restrict__ pacc,
        float* __restrict__ cpacc, float* __restrict__ den) {
    int bp = blockIdx.x;          // b*P + p
    int b = bp / P;
    int p = bp - b * P;
    int c = threadIdx.x;          // channel
    float acc = 0.0f, accw = 0.0f;
    for (int n = 0; n < NR; n++) {
        if ((n & 31) == 0 && n + 32 < NR) {
            // pull next chunk toward the caches (global_prefetch_b8)
            __builtin_prefetch(&Amat[((size_t)b * NR + n + 32) * P + p], 0, 0);
            __builtin_prefetch(&etk_t[((size_t)b * NR + n + 32) * C + c], 0, 0);
        }
        float a = Amat[((size_t)b * NR + n) * P + p];  // broadcast read
        acc += a * etk_t[((size_t)b * NR + n) * C + c];
        if (c < 3)       accw += a * ct_t[((size_t)b * NR + n) * 3 + c];
        else if (c == 3) accw += a;
    }
    pacc[(size_t)bp * C + c] = acc;
    if (c < 3)  cpacc[(size_t)bp * 3 + c] = accw;
    if (c == 3) den[bp] = accw;
}

__global__ void k_finalize_protos(const float* __restrict__ pacc,
                                  const float* __restrict__ cpacc,
                                  const float* __restrict__ den,
                                  float* __restrict__ protos, float* __restrict__ cp) {
    int bp = blockIdx.x * blockDim.x + threadIdx.x;
    if (bp >= B * P) return;
    float inv = 1.0f / den[bp];
    float ss = 0.0f;
    for (int c = 0; c < C; c++) {
        float v = pacc[(size_t)bp * C + c] * inv;
        ss += v * v;
    }
    float rn = 1.0f / fmaxf(sqrtf(ss), 1e-12f);
    for (int c = 0; c < C; c++)
        protos[(size_t)bp * C + c] = pacc[(size_t)bp * C + c] * inv * rn;
    for (int k = 0; k < 3; k++)
        cp[(size_t)bp * 3 + k] = cpacc[(size_t)bp * 3 + k] * inv;
}

// ------------------------------ loss reduction ------------------------------
__global__ void k_zero2(float* __restrict__ acc) {
    if (threadIdx.x < 2 && blockIdx.x == 0) acc[threadIdx.x] = 0.0f;
}

__global__ void k_loss(const float* __restrict__ lt, const float* __restrict__ ls,
                       const int* __restrict__ idxc, const float* __restrict__ mask,
                       float* __restrict__ acc) {
    int idx = blockIdx.x * blockDim.x + threadIdx.x; // b*NR + ns
    if (idx >= B * NR) return;
    int b = idx / NR;
    float m = mask[idx];
    int j = idxc[idx];
    const float* trow = lt + ((size_t)b * NR + j) * P;
    const float* srow = ls + (size_t)idx * P;
    float tz[P], sz[P];
    float mt = -3.4e38f, ms = -3.4e38f;
    for (int p = 0; p < P; p++) {
        tz[p] = trow[p] / TEMP_T; mt = fmaxf(mt, tz[p]);
        sz[p] = srow[p] / TEMP_S; ms = fmaxf(ms, sz[p]);
    }
    float st = 0.0f, ssum = 0.0f;
    for (int p = 0; p < P; p++) {
        tz[p] = __expf(tz[p] - mt); st += tz[p];
        ssum += __expf(sz[p] - ms);
    }
    float invt = 1.0f / st;
    float lse = logf(ssum);
    float ce = 0.0f;
    for (int p = 0; p < P; p++)
        ce -= (tz[p] * invt) * (sz[p] - ms - lse);
    atomicAdd(&acc[0], ce * m);
    atomicAdd(&acc[1], m);
}

__global__ void k_final(const float* __restrict__ acc, float* __restrict__ out) {
    if (threadIdx.x == 0 && blockIdx.x == 0)
        out[0] = acc[0] / fmaxf(acc[1], 1.0f); // LOSS_WEIGHT = 1
}

// ============================= host launcher ================================
extern "C" void kernel_launch(void* const* d_in, const int* in_sizes, int n_in,
                              void* d_out, int out_size, void* d_ws, size_t ws_size,
                              hipStream_t stream) {
    const float* es_in = (const float*)d_in[0];
    const float* et_in = (const float*)d_in[1];
    const float* cgs   = (const float*)d_in[2];
    const float* cgt   = (const float*)d_in[3];
    // frames (d_in[4]) unused: single frame -> all entries valid

    char* w = (char*)d_ws;
    size_t off = 0;
    auto alloc = [&](size_t bytes) -> void* {
        off = (off + 255) & ~(size_t)255;
        void* p = w + off;
        off += bytes;
        return p;
    };
    float* es_r    = (float*)alloc(sizeof(float) * B * C * NR);
    float* et_r    = (float*)alloc(sizeof(float) * B * C * NR);
    float* emb_p   = (float*)alloc(sizeof(float) * B * C * P);
    float* cs_r    = (float*)alloc(sizeof(float) * B * 3 * NR);
    float* ct_r    = (float*)alloc(sizeof(float) * B * 3 * NR);
    float* coord_p = (float*)alloc(sizeof(float) * B * 3 * P);
    float* etk_t   = (float*)alloc(sizeof(float) * B * NR * C);
    float* etn     = (float*)alloc(sizeof(float) * B * NR * C);
    float* esn     = (float*)alloc(sizeof(float) * B * NR * C);
    float* cs_t    = (float*)alloc(sizeof(float) * B * NR * 3);
    float* ct_t    = (float*)alloc(sizeof(float) * B * NR * 3);
    float* protos  = (float*)alloc(sizeof(float) * B * P * C);
    float* cp      = (float*)alloc(sizeof(float) * B * P * 3);
    _Float16* apT  = (_Float16*)alloc(sizeof(_Float16) * B * MT * KST * 32 * 16);
    _Float16* apS  = (_Float16*)alloc(sizeof(_Float16) * B * MT * KST * 32 * 16);
    _Float16* bpk  = (_Float16*)alloc(sizeof(_Float16) * B * NT2 * KST * 32 * 16);
    float* logits_t = (float*)alloc(sizeof(float) * B * NR * P);
    float* logits_s = (float*)alloc(sizeof(float) * B * NR * P);
    float* Amat     = (float*)alloc(sizeof(float) * B * NR * P);
    float* pacc     = (float*)alloc(sizeof(float) * B * P * C);
    float* cpacc    = (float*)alloc(sizeof(float) * B * P * 3);
    float* den      = (float*)alloc(sizeof(float) * B * P);
    int*   idxc     = (int*)alloc(sizeof(int) * B * NR);
    float* maskb    = (float*)alloc(sizeof(float) * B * NR);
    float* accum    = (float*)alloc(sizeof(float) * 2);
    (void)ws_size;

    auto blks = [](int total, int bs) { return (total + bs - 1) / bs; };

    // ---- resizes
    k_resize<<<blks(B * C * NR, 256), 256, 0, stream>>>(es_in, es_r, B * C, HR, WR, DR);
    k_resize<<<blks(B * C * NR, 256), 256, 0, stream>>>(et_in, et_r, B * C, HR, WR, DR);
    k_resize<<<blks(B * C * P, 256), 256, 0, stream>>>(et_in, emb_p, B * C, HP, WP, DP);
    k_resize<<<blks(B * 3 * NR, 256), 256, 0, stream>>>(cgs, cs_r, B * 3, HR, WR, DR);
    k_resize<<<blks(B * 3 * NR, 256), 256, 0, stream>>>(cgt, ct_r, B * 3, HR, WR, DR);
    k_resize<<<blks(B * 3 * P, 256), 256, 0, stream>>>(cgt, coord_p, B * 3, HP, WP, DP);

    // ---- transposes / normalization / proto init
    k_transpose_norm<<<blks(B * NR, 256), 256, 0, stream>>>(et_r, etk_t, etn, NR);
    k_transpose_norm<<<blks(B * NR, 256), 256, 0, stream>>>(es_r, nullptr, esn, NR);
    k_transpose_coords<<<blks(B * NR, 256), 256, 0, stream>>>(cs_r, cs_t, NR);
    k_transpose_coords<<<blks(B * NR, 256), 256, 0, stream>>>(ct_r, ct_t, NR);
    k_protos_init<<<blks(B * P, 64), 64, 0, stream>>>(emb_p, coord_p, protos, cp);

    // ---- nearest teacher voxel + mask (async-LDS staging)
    {
        dim3 g(NR / 256, B);
        k_nearest<<<g, 256, 0, stream>>>(cs_t, ct_t, idxc, maskb);
    }

    // ---- pack WMMA A operands (fixed across iterations)
    k_packA<<<blks(B * MT * KST * 32 * 16, 256), 256, 0, stream>>>(etn, apT);
    k_packA<<<blks(B * MT * KST * 32 * 16, 256), 256, 0, stream>>>(esn, apS);

    // ---- 3 k-means iterations
    for (int it = 0; it < 3; it++) {
        k_packB<<<blks(B * NT2 * KST * 32 * 16, 256), 256, 0, stream>>>(protos, bpk);
        {
            dim3 g(MT, B);
            k_gemm_wmma<<<g, 32, 0, stream>>>(apT, bpk, logits_t);
        }
        k_soft_weight<<<blks(B * NR, 256), 256, 0, stream>>>(logits_t, ct_t, cp, Amat);
        k_accum<<<B * P, 64, 0, stream>>>(Amat, etk_t, ct_t, pacc, cpacc, den);
        k_finalize_protos<<<1, 64, 0, stream>>>(pacc, cpacc, den, protos, cp);
    }

    // ---- final logits (teacher targets + student) via WMMA
    k_packB<<<blks(B * NT2 * KST * 32 * 16, 256), 256, 0, stream>>>(protos, bpk);
    {
        dim3 g(MT, B);
        k_gemm_wmma<<<g, 32, 0, stream>>>(apT, bpk, logits_t);
        k_gemm_wmma<<<g, 32, 0, stream>>>(apS, bpk, logits_s);
    }

    // ---- masked cross-entropy -> scalar
    k_zero2<<<1, 32, 0, stream>>>(accum);
    k_loss<<<blks(B * NR, 256), 256, 0, stream>>>(logits_t, logits_s, idxc, maskb, accum);
    k_final<<<1, 32, 0, stream>>>(accum, (float*)d_out);
}
